// DKVMN_18657337934641
// MI455X (gfx1250) — compile-verified, matrix operand order checked by reading, not run on hardware
//
#include <hip/hip_runtime.h>
#include <hip/hip_bf16.h>

#define BB 64
#define SS 200
#define DK 128
#define SLOTS 50
#define NUMC 500
#define MAXC 4
#define NTOK (BB * SS)      // 12800
#define NTILE (NTOK / 16)   // 800

typedef __attribute__((ext_vector_type(16))) __bf16 bf16x16;
typedef __attribute__((ext_vector_type(8)))  float  v8f;
typedef unsigned short u16;

__device__ __forceinline__ u16 f2bf(float f) {
    unsigned int x = __float_as_uint(f);
    unsigned int r = x + 0x7FFFu + ((x >> 16) & 1u);   // RNE
    return (u16)(r >> 16);
}

__device__ __forceinline__ float fast_sigmoid(float x) {
    // 1/(1+e^-x) via v_exp + v_rcp instead of the IEEE divide sequence
    return __builtin_amdgcn_rcpf(1.0f + __expf(-x));
}

__device__ __forceinline__ float fast_tanh(float x) {
#if __has_builtin(__builtin_amdgcn_tanhf)
    return __builtin_amdgcn_tanhf(x);
#else
    return tanhf(x);
#endif
}

// Load one 16-element bf16 operand vector for WMMA A/B (16-bit striping:
// lane group g selects K offsets [kbase+g*8, +8) and [kbase+16+g*8, +8)).
__device__ __forceinline__ bf16x16 ld16(const u16* row, int kbase, int g) {
    union { bf16x16 v; uint4 q[2]; } u;
    u.q[0] = *(const uint4*)(row + kbase + g * 8);
    u.q[1] = *(const uint4*)(row + kbase + 16 + g * 8);
    return u.v;
}

// ---------------- K0: weights -> bf16 (Mk padded 50->64 rows) ----------------
__global__ void convw_kernel(const float* Mk, const float* eW, const float* aW,
                             const float* fW, u16* MkP, u16* eWb, u16* aWb, u16* fWb) {
    int i = blockIdx.x * blockDim.x + threadIdx.x;
    if (i < 64 * DK) {
        int r = i >> 7;
        MkP[i] = f2bf(r < SLOTS ? Mk[i] : 0.0f);
    }
    if (i < DK * DK) {
        eWb[i] = f2bf(eW[i]);
        aWb[i] = f2bf(aW[i]);
    }
    if (i < DK * 2 * DK) {
        fWb[i] = f2bf(fW[i]);
    }
}

// ---------------- K1: embedding gather + masked mean -> k_bf, v_bf -----------
__global__ void gather_kernel(const int* qseq, const int* cseq,
                              const int* q2c, const int* q2cm,
                              const float* keyE, const float* valE,
                              u16* kbf, u16* vbf) {
    int token = blockIdx.x;
    int d = threadIdx.x;
    int q = qseq[token];
    int corr = cseq[token];
    float ks = 0.f, vs = 0.f, cnt = 0.f;
#pragma unroll
    for (int c = 0; c < MAXC; ++c) {
        int cid = q2c[q * MAXC + c];
        float m = (float)q2cm[q * MAXC + c];
        cnt += m;
        ks += m * keyE[cid * DK + d];
        vs += m * valE[(cid + NUMC * corr) * DK + d];
    }
    float inv = __builtin_amdgcn_rcpf(fmaxf(cnt, 1.0f));
    kbf[token * DK + d] = f2bf(ks * inv);
    vbf[token * DK + d] = f2bf(vs * inv);
}

// ---------------- K2: WMMA projections: w_logits, e=sigmoid, a=tanh ----------
__global__ void proj_kernel(const u16* __restrict__ kbf, const u16* __restrict__ vbf,
                            const u16* __restrict__ MkP, const u16* __restrict__ eWb,
                            const u16* __restrict__ aWb,
                            const float* __restrict__ e_b, const float* __restrict__ a_b,
                            float* __restrict__ e_out, float* __restrict__ a_out,
                            float* __restrict__ wlog) {
    int wid = threadIdx.x >> 5, lane = threadIdx.x & 31;
    int tile = blockIdx.x * 8 + wid;
    int g = lane >> 4, c = lane & 15;
    int rowTok = tile * 16 + c;

    // Warm WGP$ with the (small, L2-resident) weight matrices.
    __builtin_prefetch(eWb + lane * 256, 0, 3);
    __builtin_prefetch(aWb + lane * 256, 0, 3);
    __builtin_prefetch(MkP + lane * 256, 0, 3);

    // ---- Phase 1: A = v  (e and a projections) ----
    bf16x16 Av[4];
#pragma unroll
    for (int kk = 0; kk < 4; ++kk)
        Av[kk] = ld16(vbf + rowTok * DK, kk * 32, g);

    // e = sigmoid(v @ eW^T + e_b)
#pragma unroll
    for (int nb = 0; nb < 8; ++nb) {
        v8f acc = {};
#pragma unroll
        for (int kk = 0; kk < 4; ++kk) {
            bf16x16 Bm = ld16(eWb + (nb * 16 + c) * DK, kk * 32, g);
            acc = __builtin_amdgcn_wmma_f32_16x16x32_bf16(false, Av[kk], false, Bm,
                                                          (short)0, acc, false, false);
        }
        int col = nb * 16 + c;
        float bias = e_b[col];
#pragma unroll
        for (int i = 0; i < 8; ++i) {
            int tok = tile * 16 + i + 8 * g;
            e_out[tok * DK + col] = fast_sigmoid(acc[i] + bias);
        }
    }
    // a = tanh(v @ aW^T + a_b)
#pragma unroll
    for (int nb = 0; nb < 8; ++nb) {
        v8f acc = {};
#pragma unroll
        for (int kk = 0; kk < 4; ++kk) {
            bf16x16 Bm = ld16(aWb + (nb * 16 + c) * DK, kk * 32, g);
            acc = __builtin_amdgcn_wmma_f32_16x16x32_bf16(false, Av[kk], false, Bm,
                                                          (short)0, acc, false, false);
        }
        int col = nb * 16 + c;
        float bias = a_b[col];
#pragma unroll
        for (int i = 0; i < 8; ++i) {
            int tok = tile * 16 + i + 8 * g;
            a_out[tok * DK + col] = fast_tanh(acc[i] + bias);
        }
    }

    // ---- Phase 2: A = k  (w logits; Mk padded to 64 rows) ----
    bf16x16 Ak[4];
#pragma unroll
    for (int kk = 0; kk < 4; ++kk)
        Ak[kk] = ld16(kbf + rowTok * DK, kk * 32, g);

#pragma unroll
    for (int nb = 0; nb < 4; ++nb) {
        v8f acc = {};
#pragma unroll
        for (int kk = 0; kk < 4; ++kk) {
            bf16x16 Bm = ld16(MkP + (nb * 16 + c) * DK, kk * 32, g);
            acc = __builtin_amdgcn_wmma_f32_16x16x32_bf16(false, Ak[kk], false, Bm,
                                                          (short)0, acc, false, false);
        }
        int col = nb * 16 + c;
#pragma unroll
        for (int i = 0; i < 8; ++i) {
            int tok = tile * 16 + i + 8 * g;
            wlog[tok * 64 + col] = acc[i];
        }
    }
}

// ---------------- K2b: per-token softmax over 50 slots (in-place) ------------
__global__ void softmax_kernel(float* wlog) {
    int token = blockIdx.x * blockDim.x + threadIdx.x;
    if (token >= NTOK) return;
    float* p = wlog + token * 64;
    float mx = -3.0e38f;
    for (int n = 0; n < SLOTS; ++n) mx = fmaxf(mx, p[n]);
    float s = 0.f;
    for (int n = 0; n < SLOTS; ++n) { float ev = __expf(p[n] - mx); p[n] = ev; s += ev; }
    float inv = __builtin_amdgcn_rcpf(s);
    for (int n = 0; n < SLOTS; ++n) p[n] *= inv;
}

// ---------------- K3: sequential scan over S; Mv register-resident -----------
__global__ void scan_kernel(const float* __restrict__ w, const float* __restrict__ e,
                            const float* __restrict__ a, const float* __restrict__ Mv0,
                            u16* __restrict__ readsbf) {
    __shared__ float wsl[64];
    int b = blockIdx.x, half = blockIdx.y, tid = threadIdx.x;
    int d = half * 64 + tid;

    float Mv[SLOTS];
#pragma unroll
    for (int n = 0; n < SLOTS; ++n) Mv[n] = Mv0[n * DK + d];

    for (int t = 0; t < SS; ++t) {
        int token = b * SS + t;
        if (tid < SLOTS) wsl[tid] = w[token * 64 + tid];
        float ed = e[token * DK + d];
        float ad = a[token * DK + d];
        __syncthreads();
        float rd = 0.f;
#pragma unroll
        for (int n = 0; n < SLOTS; ++n) {
            float wn = wsl[n];
            rd += wn * Mv[n];                            // read BEFORE write
            Mv[n] = Mv[n] * (1.0f - wn * ed) + wn * ad;
        }
        readsbf[token * DK + d] = f2bf(rd);
        __syncthreads();
    }
}

// ---------------- K4: f = tanh([reads|k] @ fW^T + f_b); out = sigmoid(f.pW) --
__global__ void final_kernel(const u16* __restrict__ readsbf, const u16* __restrict__ kbf,
                             const u16* __restrict__ fWb, const float* __restrict__ f_b,
                             const float* __restrict__ p_W, const float* __restrict__ p_b,
                             float* __restrict__ out) {
    int wid = threadIdx.x >> 5, lane = threadIdx.x & 31;
    int tile = blockIdx.x * 8 + wid;
    int g = lane >> 4, c = lane & 15;
    int rowTok = tile * 16 + c;

    __builtin_prefetch(fWb + lane * 1024, 0, 3);

    bf16x16 A[8];
#pragma unroll
    for (int kk = 0; kk < 8; ++kk) {
        if (kk < 4) A[kk] = ld16(readsbf + rowTok * DK, kk * 32, g);
        else        A[kk] = ld16(kbf + rowTok * DK, (kk - 4) * 32, g);
    }

    float partial[8];
#pragma unroll
    for (int i = 0; i < 8; ++i) partial[i] = 0.f;

#pragma unroll
    for (int nb = 0; nb < 8; ++nb) {
        v8f acc = {};
#pragma unroll
        for (int kk = 0; kk < 8; ++kk) {
            bf16x16 Bm = ld16(fWb + (nb * 16 + c) * (2 * DK), kk * 32, g);
            acc = __builtin_amdgcn_wmma_f32_16x16x32_bf16(false, A[kk], false, Bm,
                                                          (short)0, acc, false, false);
        }
        int col = nb * 16 + c;
        float pw = p_W[col];
        float fb = f_b[col];
#pragma unroll
        for (int i = 0; i < 8; ++i)
            partial[i] += fast_tanh(acc[i] + fb) * pw;
    }
    // reduce across the 16 lanes of each row group
#pragma unroll
    for (int off = 1; off < 16; off <<= 1)
#pragma unroll
        for (int i = 0; i < 8; ++i)
            partial[i] += __shfl_xor(partial[i], off, 32);

    if (c == 0) {
        float pb = p_b[0];
#pragma unroll
        for (int i = 0; i < 8; ++i) {
            int tok = tile * 16 + i + 8 * g;
            out[tok] = fast_sigmoid(partial[i] + pb);
        }
    }
}

extern "C" void kernel_launch(void* const* d_in, const int* in_sizes, int n_in,
                              void* d_out, int out_size, void* d_ws, size_t ws_size,
                              hipStream_t stream) {
    const int*   qseq  = (const int*)d_in[0];
    const int*   cseq  = (const int*)d_in[1];
    const int*   q2c   = (const int*)d_in[2];
    const int*   q2cm  = (const int*)d_in[3];
    const float* keyE  = (const float*)d_in[4];
    const float* valE  = (const float*)d_in[5];
    const float* Mk    = (const float*)d_in[6];
    const float* Mv0   = (const float*)d_in[7];
    const float* f_W   = (const float*)d_in[8];
    const float* f_b   = (const float*)d_in[9];
    const float* e_W   = (const float*)d_in[10];
    const float* e_b   = (const float*)d_in[11];
    const float* a_W   = (const float*)d_in[12];
    const float* a_b   = (const float*)d_in[13];
    const float* p_W   = (const float*)d_in[14];
    const float* p_b   = (const float*)d_in[15];
    float* out = (float*)d_out;

    char* base = (char*)d_ws;
    size_t off = 0;
    auto take = [&](size_t bytes) -> void* {
        void* r = base + off;
        off = (off + bytes + 255) & ~(size_t)255;
        return r;
    };
    u16*   kbf     = (u16*)  take((size_t)NTOK * DK * 2);
    u16*   vbf     = (u16*)  take((size_t)NTOK * DK * 2);
    u16*   readsbf = (u16*)  take((size_t)NTOK * DK * 2);
    u16*   MkP     = (u16*)  take((size_t)64 * DK * 2);
    u16*   eWb     = (u16*)  take((size_t)DK * DK * 2);
    u16*   aWb     = (u16*)  take((size_t)DK * DK * 2);
    u16*   fWb     = (u16*)  take((size_t)DK * 2 * DK * 2);
    float* wbuf    = (float*)take((size_t)NTOK * 64 * 4);
    float* e_f     = (float*)take((size_t)NTOK * DK * 4);
    float* a_f     = (float*)take((size_t)NTOK * DK * 4);

    convw_kernel<<<128, 256, 0, stream>>>(Mk, e_W, a_W, f_W, MkP, eWb, aWb, fWb);
    gather_kernel<<<NTOK, DK, 0, stream>>>(qseq, cseq, q2c, q2cm, keyE, valE, kbf, vbf);
    proj_kernel<<<NTILE / 8, 256, 0, stream>>>(kbf, vbf, MkP, eWb, aWb, e_b, a_b,
                                               e_f, a_f, wbuf);
    softmax_kernel<<<(NTOK + 255) / 256, 256, 0, stream>>>(wbuf);
    scan_kernel<<<dim3(BB, 2), 64, 0, stream>>>(wbuf, e_f, a_f, Mv0, readsbf);
    final_kernel<<<NTILE / 8, 256, 0, stream>>>(readsbf, kbf, fWb, f_b, p_W, p_b, out);
}
